// GraphSequential_75376676044986
// MI455X (gfx1250) — matrix operand order repaired and checked by reading, not compiled, not run on hardware
//
#include <hip/hip_runtime.h>

// ---------------------------------------------------------------------------
// EdgeConv (Linear on [x_i, x_j-x_i]) + segment-max + per-graph InstanceNorm
// + LeakyReLU, fused for MI455X (gfx1250, wave32, WMMA).
//
// Heavy kernel: per-wave 16-edge tiles -> v_wmma_f32_16x16x32_f16 (4 Ktiles x
// 4 Ntiles) -> global_atomic_max_num_f32 (saddr form) scatter into the
// L2-resident agg buffer. Atomic MAX is order-independent => deterministic.
// Bias folded in after aggregation (max(z)+b == max(z+b)).
// ---------------------------------------------------------------------------

#define N_NODES   50000
#define N_EDGES   1600000
#define C_IN      64
#define C_OUT     64
#define N_GRAPHS  16
#define NGROUPS   (N_EDGES / 16)     // 100000 exact
#define GROUPS_PER_WAVE 8
#define WAVES_PER_BLOCK 8

typedef __attribute__((ext_vector_type(16))) _Float16 v16h;
typedef __attribute__((ext_vector_type(8)))  _Float16 v8h;
typedef __attribute__((ext_vector_type(8)))  float    v8f;

// ---------------------------------------------------------------------------
// 1) init aggregation buffer to -inf
// ---------------------------------------------------------------------------
__global__ void init_agg_kernel(float* __restrict__ agg) {
  int i = blockIdx.x * blockDim.x + threadIdx.x;
  if (i < N_NODES * C_OUT) agg[i] = -__builtin_huge_valf();
}

// ---------------------------------------------------------------------------
// 2) pack W (128x64 f32, row-major) into f16 WMMA B fragments.
// Fragment f = kt*4 + nt covers K in [kt*32, kt*32+32), N in [nt*16, nt*16+16).
// B lane order used here (contiguous K, 32x16): lane%16 = column N; lanes 0-15 hold
// K 0..15 (VGPR r = K {2r,2r+1}), lanes 16-31 hold K 16..31.
// packedW[((f*32 + lane)*16 + elem)]  (v16h per lane per fragment)
// ---------------------------------------------------------------------------
__global__ void pack_w_kernel(const float* __restrict__ W,
                              _Float16* __restrict__ packedW) {
  int idx = blockIdx.x * blockDim.x + threadIdx.x;   // 0..8191
  if (idx >= 16 * 32 * 16) return;
  int e = idx & 15;
  int l = (idx >> 4) & 31;
  int f = idx >> 9;
  int kt = f >> 2, nt = f & 3;
  int r = e >> 1, pair = e & 1;
  int klocal = ((l < 16) ? 0 : 16) + 2 * r + pair;
  int k = kt * 32 + klocal;
  int n = nt * 16 + (l & 15);
  packedW[idx] = (_Float16)W[k * C_OUT + n];
}

// ---------------------------------------------------------------------------
// 3) fused edge GEMM + atomic-max scatter
// ---------------------------------------------------------------------------
__global__ __launch_bounds__(256) void edge_wmma_kernel(
    const float* __restrict__ x, const long long* __restrict__ ei,
    const _Float16* __restrict__ packedW, float* __restrict__ agg) {
  __shared__ _Float16 ldsA[WAVES_PER_BLOCK][16 * 128];  // 16 edges x 128 halfs
  __shared__ alignas(16) int ldsDst[WAVES_PER_BLOCK][16];

  const int lane = threadIdx.x & 31;
  const int wave = threadIdx.x >> 5;
  const long long* srcIdx = ei;             // edge_index row 0 = src (j)
  const long long* dstIdx = ei + N_EDGES;   // edge_index row 1 = dst (i)

  // W fragments live in registers for the whole wave (reused by every edge).
  v16h Bf[16];
  const v16h* wp = (const v16h*)packedW;
#pragma unroll
  for (int f = 0; f < 16; ++f) Bf[f] = wp[f * 32 + lane];

  const int groupBase = (blockIdx.x * WAVES_PER_BLOCK + wave) * GROUPS_PER_WAVE;

  for (int gi = 0; gi < GROUPS_PER_WAVE; ++gi) {
    const int  group  = groupBase + gi;
    const bool active = group < NGROUPS;
    const int  cg     = active ? group : (NGROUPS - 1);

    // ---- stage A tile: 16 edges x [x_i | x_j - x_i] as f16 in LDS ----
    {
      int m   = lane & 15;   // edge within group
      int sel = lane >> 4;   // channel half: 0 -> 0..31, 1 -> 32..63
      long long e = (long long)cg * 16 + m;
      long long s = srcIdx[e];
      long long d = dstIdx[e];
      if (sel == 0) ldsDst[wave][m] = (int)d;
      const float4* xi4 = (const float4*)(x + d * C_IN + sel * 32);
      const float4* xj4 = (const float4*)(x + s * C_IN + sel * 32);
      _Float16* ai = &ldsA[wave][m * 128 + sel * 32];
      _Float16* ad = ai + 64;
#pragma unroll
      for (int q = 0; q < 8; ++q) {
        float4 vi = xi4[q], vj = xj4[q];
        ai[q * 4 + 0] = (_Float16)vi.x;
        ai[q * 4 + 1] = (_Float16)vi.y;
        ai[q * 4 + 2] = (_Float16)vi.z;
        ai[q * 4 + 3] = (_Float16)vi.w;
        ad[q * 4 + 0] = (_Float16)(vj.x - vi.x);
        ad[q * 4 + 1] = (_Float16)(vj.y - vi.y);
        ad[q * 4 + 2] = (_Float16)(vj.z - vi.z);
        ad[q * 4 + 3] = (_Float16)(vj.w - vi.w);
      }
    }
    __syncthreads();

    // ---- preload all A fragments (one dscnt wait, then WMMAs back-to-back)
    v16h Af[4];
    {
      int row = lane & 15;
      int kb  = (lane < 16) ? 0 : 16;  // A order used here: lane half selects K half (contiguous, as packed B)
      const _Float16* Ap = &ldsA[wave][row * 128];
#pragma unroll
      for (int kt = 0; kt < 4; ++kt) {
        v8h lo = *(const v8h*)(Ap + kt * 32 + kb);        // K kb..kb+7
        v8h hi = *(const v8h*)(Ap + kt * 32 + kb + 8);    // K kb+8..kb+15
#pragma unroll
        for (int i2 = 0; i2 < 8; ++i2) {
          Af[kt][i2]     = lo[i2];
          Af[kt][8 + i2] = hi[i2];
        }
      }
    }

    // ---- 16x64 = A(16x128) * W(128x64): 4 Ktiles x 4 Ntiles of WMMA ----
    v8f acc[4] = {};
#pragma unroll
    for (int kt = 0; kt < 4; ++kt) {
#pragma unroll
      for (int nt = 0; nt < 4; ++nt) {
        acc[nt] = __builtin_amdgcn_wmma_f32_16x16x32_f16(
            false, Af[kt], false, Bf[kt * 4 + nt], (short)0, acc[nt],
            false, false);
      }
    }

    // ---- scatter with CDNA5 float-max atomics (saddr form, imm offsets) ----
    if (active) {
      int col  = lane & 15;            // C/D layout: lane%16 = column N
      int mofs = (lane >> 4) * 8;      // lanes >=16 hold rows M+8
      const int4* dp = (const int4*)&ldsDst[wave][mofs];
      int4 d0 = dp[0];
      int4 d1 = dp[1];
      int drow[8] = {d0.x, d0.y, d0.z, d0.w, d1.x, d1.y, d1.z, d1.w};
#pragma unroll
      for (int r = 0; r < 8; ++r) {
        // byte offset of (node drow[r], channel col) within agg
        unsigned voff = ((unsigned)drow[r] << 8) + ((unsigned)col << 2);
#pragma unroll
        for (int nt = 0; nt < 4; ++nt) {
          asm volatile("global_atomic_max_num_f32 %0, %1, %2 offset:%3"
                       :: "v"(voff), "v"(acc[nt][r]), "s"(agg),
                          "i"(nt * 16 * 4)
                       : "memory");
        }
      }
    }
    __syncthreads();
  }
}

// ---------------------------------------------------------------------------
// 4) graph boundaries from sorted batch (one writer per slot -> deterministic)
// ---------------------------------------------------------------------------
__global__ void bounds_kernel(const long long* __restrict__ batch,
                              int* __restrict__ starts) {
  int n = blockIdx.x * blockDim.x + threadIdx.x;
  if (n >= N_NODES) return;
  int g  = (int)batch[n];
  int gp = (n == 0) ? -1 : (int)batch[n - 1];
  for (int gg = gp + 1; gg <= g; ++gg) starts[gg] = n;
  if (n == N_NODES - 1)
    for (int gg = g + 1; gg <= N_GRAPHS; ++gg) starts[gg] = N_NODES;
}

// ---------------------------------------------------------------------------
// 5) per-(graph,channel) biased mean/var (fixed-order reduce -> deterministic)
//    applies the  (-inf -> 0, +bias)  transform while reading agg
// ---------------------------------------------------------------------------
__global__ __launch_bounds__(256) void stats_kernel(
    const float* __restrict__ agg, const int* __restrict__ starts,
    const float* __restrict__ bias, float* __restrict__ meanArr,
    float* __restrict__ scaleArr) {
  __shared__ float ssum[4][64];
  __shared__ float ssq[4][64];
  int g = blockIdx.x;
  int s = starts[g], e = starts[g + 1];
  float cnt = fmaxf((float)(e - s), 1.0f);
  int c = threadIdx.x & 63, r = threadIdx.x >> 6;
  float bc = bias[c];
  float sum = 0.f, sq = 0.f;
  for (int n = s + r; n < e; n += 4) {
    float a = agg[(size_t)n * C_OUT + c];
    float v = (a == -__builtin_huge_valf()) ? 0.f : a + bc;
    sum += v;
    sq  += v * v;
  }
  ssum[r][c] = sum;
  ssq[r][c]  = sq;
  __syncthreads();
  if (r == 0) {
    float S = ssum[0][c] + ssum[1][c] + ssum[2][c] + ssum[3][c];
    float Q = ssq[0][c] + ssq[1][c] + ssq[2][c] + ssq[3][c];
    float mean = S / cnt;
    float var  = fmaxf(Q / cnt - mean * mean, 0.f);
    meanArr[g * C_OUT + c]  = mean;
    scaleArr[g * C_OUT + c] = rsqrtf(var + 1e-5f);
  }
}

// ---------------------------------------------------------------------------
// 6) normalize + LeakyReLU(0.01)
// ---------------------------------------------------------------------------
__global__ void final_kernel(const float* __restrict__ agg,
                             const long long* __restrict__ batch,
                             const float* __restrict__ bias,
                             const float* __restrict__ meanArr,
                             const float* __restrict__ scaleArr,
                             float* __restrict__ out) {
  int idx = blockIdx.x * blockDim.x + threadIdx.x;
  if (idx >= N_NODES * C_OUT) return;
  int n = idx >> 6, c = idx & 63;
  int g = (int)batch[n];
  float a = agg[idx];
  float v = (a == -__builtin_huge_valf()) ? 0.f : a + bias[c];
  float o = (v - meanArr[g * C_OUT + c]) * scaleArr[g * C_OUT + c];
  out[idx] = (o >= 0.f) ? o : 0.01f * o;
}

// ---------------------------------------------------------------------------
extern "C" void kernel_launch(void* const* d_in, const int* in_sizes, int n_in,
                              void* d_out, int out_size, void* d_ws,
                              size_t ws_size, hipStream_t stream) {
  (void)in_sizes; (void)n_in; (void)out_size; (void)ws_size;
  const float*      x     = (const float*)d_in[0];
  const long long*  ei    = (const long long*)d_in[1];  // int64 (2, E)
  const long long*  batch = (const long long*)d_in[2];  // int64 (N)
  const float*      W     = (const float*)d_in[3];      // (128, 64)
  const float*      b     = (const float*)d_in[4];      // (64,)
  float*            out   = (float*)d_out;

  // workspace layout (all 16B-aligned)
  float*    agg      = (float*)d_ws;                       // 3.2M f32 = 12.8MB
  _Float16* packedW  = (_Float16*)(agg + N_NODES * C_OUT); // 8192 f16
  int*      starts   = (int*)(packedW + 8192);             // 17 (padded 32)
  float*    meanArr  = (float*)(starts + 32);              // 1024 f32
  float*    scaleArr = meanArr + N_GRAPHS * C_OUT;         // 1024 f32

  init_agg_kernel<<<(N_NODES * C_OUT + 255) / 256, 256, 0, stream>>>(agg);
  pack_w_kernel<<<32, 256, 0, stream>>>(W, packedW);
  bounds_kernel<<<(N_NODES + 255) / 256, 256, 0, stream>>>(batch, starts);

  int blocks =
      (NGROUPS + WAVES_PER_BLOCK * GROUPS_PER_WAVE - 1) /
      (WAVES_PER_BLOCK * GROUPS_PER_WAVE);
  edge_wmma_kernel<<<blocks, 256, 0, stream>>>(x, ei, packedW, agg);

  stats_kernel<<<N_GRAPHS, 256, 0, stream>>>(agg, starts, b, meanArr, scaleArr);
  final_kernel<<<(N_NODES * C_OUT + 255) / 256, 256, 0, stream>>>(
      agg, batch, b, meanArr, scaleArr, out);
}
